// kanConv2d_61598420959830
// MI455X (gfx1250) — matrix operand order repaired and checked by reading, not compiled, not run on hardware
//
#include <hip/hip_runtime.h>
#include <hip/hip_bf16.h>
#include <math.h>

// ---------------------------------------------------------------------------
// KAN-Conv2d fused as one GEMM:  y[M,64] = A[M,1152] * Wc[1152,64]
//   M = B*Ho*Wo = 32*62*62 = 123008,  D = C*K*K = 144
//   inner dim: per input-feature d, 8 slots = {6 B-spline bases, silu, 0}
//   Wc[d*8+n][o] = coef[d][o][n]*scale_sp[d][o] (n<6) | scale_base[d][o] (n==6) | 0
// WMMA: v_wmma_f32_16x16x32_f16, transposed tiles (A-op = W^T, B-op = Act^T)
// so D-fragment lanes map to contiguous wo -> coalesced stores.
// ---------------------------------------------------------------------------

typedef __attribute__((ext_vector_type(16))) _Float16 v16h;
typedef __attribute__((ext_vector_type(8)))  _Float16 v8h;
typedef __attribute__((ext_vector_type(8)))  float    v8f;

#define CIN    16
#define HIN    64
#define WIN    64
#define BATCH  32
#define DIN    144          // CIN*3*3
#define OUTD   64
#define HO     62
#define WO     62
#define PIX    (HO*WO)      // 3844
#define MTOT   (BATCH*PIX)  // 123008
#define KD     1152         // DIN*8, padded inner dim (36 chunks of 32)
#define NCHUNK 36
#define ROWS   32           // M-rows per workgroup
#define LDSS   1160         // fp16 elems per LDS row (1152 + 8 pad -> 4-bank skew)

// --- uniform-grid cubic B-spline basis (G=3, k=3, grid on [-1,1]) ----------
__device__ __forceinline__ void bspline6(float x, float bs[6]) {
    const float h = 2.0f / 3.0f;
    float t[10];
#pragma unroll
    for (int j = 0; j < 10; ++j) t[j] = (float)(j - 3) * h - 1.0f;
    float b[9];
#pragma unroll
    for (int j = 0; j < 9; ++j) b[j] = (x >= t[j] && x < t[j + 1]) ? 1.0f : 0.0f;
#pragma unroll
    for (int d = 1; d <= 3; ++d) {
        const float inv = 1.0f / ((float)d * h);   // uniform knots: t[j+d]-t[j] = d*h
#pragma unroll
        for (int j = 0; j < 9 - d; ++j)
            b[j] = ((x - t[j]) * b[j] + (t[j + d + 1] - x) * b[j + 1]) * inv;
    }
#pragma unroll
    for (int n = 0; n < 6; ++n) bs[n] = b[n];
}

// --- Kernel 1: pack Wc into lane-swizzled fp16 A-fragment layout -----------
// wpack[((c*4+ob)*32+lane)*16 + i] = Wc[k][o]
//   k = c*32 + (i<8 ? i : i+8) + (lane>=16 ? 8 : 0)   (ISA 16-bit A-matrix map)
//   o = ob*16 + (lane&15)
__global__ void kan_pack_w(const float* __restrict__ coef,
                           const float* __restrict__ scale_base,
                           const float* __restrict__ scale_sp,
                           _Float16* __restrict__ wpack) {
    const int blk  = blockIdx.x;          // 0..143 = c*4 + ob
    const int lane = threadIdx.x;         // 0..31
    const int c    = blk >> 2;
    const int ob   = blk & 3;
    const int o    = ob * 16 + (lane & 15);
    const int khi  = (lane >= 16) ? 8 : 0;

    _Float16* dst = wpack + ((size_t)(blk * 32 + lane) << 4);
#pragma unroll
    for (int i = 0; i < 16; ++i) {
        const int k = c * 32 + ((i < 8) ? i : i + 8) + khi;
        const int d = k >> 3;
        const int n = k & 7;
        float w;
        if      (n < 6)  w = coef[(d * OUTD + o) * 6 + n] * scale_sp[d * OUTD + o];
        else if (n == 6) w = scale_base[d * OUTD + o];
        else             w = 0.0f;
        dst[i] = (_Float16)w;
    }
}

// --- Kernel 2: fused activation-gen + WMMA GEMM ----------------------------
__global__ __launch_bounds__(256) void kan_main(const float* __restrict__ x,
                                                const _Float16* __restrict__ wpack,
                                                float* __restrict__ y) {
    extern __shared__ char smem_raw[];
    _Float16* smem = (_Float16*)smem_raw;          // [ROWS][LDSS] fp16

    const int tid = threadIdx.x;
    const int wg  = blockIdx.x;

    // ---------------- Phase 1: build activation tile Act[32][1152] --------
    {
        const int rowL = tid >> 3;                 // 0..31
        const int dOff = tid & 7;                  // 8 threads per row
        const int m    = wg * ROWS + rowL;
        const int b    = m / PIX;
        const int rem  = m - b * PIX;
        const int ho   = rem / WO;
        const int wo   = rem - ho * WO;
        const float* xb = x + (size_t)b * (CIN * HIN * WIN);

#pragma unroll 2
        for (int j = 0; j < DIN / 8; ++j) {        // 18 iterations
            const int d  = dOff + (j << 3);
            const int ci = d / 9;
            const int rr = d - ci * 9;
            const int di = rr / 3;
            const int dj = rr - di * 3;
            const float p = xb[(ci * HIN + ho + di) * WIN + (wo + dj)];

            float bs[6];
            bspline6(p, bs);
            const float sil = p * (1.0f / (1.0f + __expf(-p)));

            v8h pk;
#pragma unroll
            for (int n = 0; n < 6; ++n) pk[n] = (_Float16)bs[n];
            pk[6] = (_Float16)sil;
            pk[7] = (_Float16)0.0f;
            *(v8h*)(smem + rowL * LDSS + (d << 3)) = pk;   // 16B aligned
        }
    }
    __syncthreads();

    // ---------------- Phase 2: one 16(o) x 16(m) tile per wave -------------
    const int lane = tid & 31;
    const int wave = tid >> 5;                     // 0..7
    const int l16  = lane & 15;
    const int hi   = lane >> 4;                    // 0/1 half-wave
    const int mt   = wave >> 2;                    // 0..1  row tile (16 m each)
    const int ob   = wave & 3;                     // 0..3  col tile (16 o each)

    const int brow = mt * 16 + l16;                // activation row (m) this lane serves
    const _Float16* bbase = smem + brow * LDSS + hi * 8;
    const v16h* abase = (const v16h*)wpack + (size_t)ob * 32 + lane;

    v8f acc = {};
#pragma unroll 4
    for (int c = 0; c < NCHUNK; ++c) {
        if (c + 1 < NCHUNK)
            __builtin_prefetch((const void*)(abase + (size_t)(c + 1) * 128), 0, 1);

        // A-fragment: weights W^T chunk, 32 contiguous bytes per lane
        const v16h a = abase[(size_t)c * 128];     // stride = 4 ob-blocks * 32 lanes

        // B-fragment: activations^T from LDS, lane -> (m=l16, k = 8*hi+i, 16+8*hi+i)
        const v8h lo = *(const v8h*)(bbase + c * 32);
        const v8h hi8 = *(const v8h*)(bbase + c * 32 + 16);
        v16h bf;
#pragma unroll
        for (int i = 0; i < 8; ++i) { bf[i] = lo[i]; bf[i + 8] = hi8[i]; }

        acc = __builtin_amdgcn_wmma_f32_16x16x32_f16(
            /*neg_a=*/false, a, /*neg_b=*/false, bf,
            /*c_mod=*/(short)0, acc, /*reuse_a=*/false, /*reuse_b=*/false);
    }

    // ---------------- Store: D lanes = m (contiguous wo), VGPR r = o -------
    {
        const int mg  = wg * ROWS + mt * 16 + l16;
        const int b   = mg / PIX;
        const int rem = mg - b * PIX;
        const int ho  = rem / WO;
        const int wo  = rem - ho * WO;
        float* ybase = y + ((size_t)b * OUTD) * PIX + ho * WO + wo;
        const int o0 = ob * 16 + hi * 8;
#pragma unroll
        for (int r = 0; r < 8; ++r)
            ybase[(size_t)(o0 + r) * PIX] = acc[r];
    }
}

extern "C" void kernel_launch(void* const* d_in, const int* in_sizes, int n_in,
                              void* d_out, int out_size, void* d_ws, size_t ws_size,
                              hipStream_t stream) {
    const float* x          = (const float*)d_in[0];   // [32,16,64,64]
    const float* coef       = (const float*)d_in[1];   // [144,64,6]
    const float* scale_base = (const float*)d_in[2];   // [144,64]
    const float* scale_sp   = (const float*)d_in[3];   // [144,64]
    float*       y          = (float*)d_out;           // [32,64,62,62]
    _Float16*    wpack      = (_Float16*)d_ws;         // 36*4*32*16 fp16 = 144 KB

    kan_pack_w<<<NCHUNK * 4, 32, 0, stream>>>(coef, scale_base, scale_sp, wpack);

    const size_t lds_bytes = (size_t)ROWS * LDSS * sizeof(_Float16); // 74240 B
    kan_main<<<MTOT / ROWS, 256, lds_bytes, stream>>>(x, wpack, y);
}
